// ScaledDotProductAttention_1331439862428
// MI455X (gfx1250) — compile-verified, matrix operand order checked by reading, not run on hardware
//
#include <hip/hip_runtime.h>
#include <cstdint>
#include <cstddef>

typedef __attribute__((ext_vector_type(16))) _Float16 v16h;
typedef __attribute__((ext_vector_type(8)))  float    v8f;
typedef __attribute__((ext_vector_type(4)))  float    v4f;
typedef __attribute__((ext_vector_type(4)))  _Float16 v4h;
typedef __attribute__((ext_vector_type(2)))  _Float16 v2h;

#define B_    32
#define L_    2048
#define D_    64
#define SCALE 0.125f              // 1/temperature, temperature = 8
#define MASKH ((_Float16)(-60000.0f))  // f16-representable mask fill

// LDS strides (halfs) padded for conflict-free 16-lane row gathers
#define P_STRIDE 2056             // 16 rows of scores/probs, f16
#define Q_STRIDE 72               // 16 rows of Q, f16

#define SMEM_P   (16 * P_STRIDE * 2)
#define SMEM_Q   (16 * Q_STRIDE * 2)
#define SMEM_O   (2 * 16 * 64 * 4)
#define SMEM_INV (16 * 4)
#define SMEM_TOTAL (SMEM_P + SMEM_Q + SMEM_O + SMEM_INV)

// ISA 7.12.2: 16-bit A/B fragment, lane-half h, packed pair j (0..7) starts at
// K = 2j + 8*(j>=4) + 8h  (covers K and K+1)
__device__ __forceinline__ int frag_d(int j, int h) {
    return 2 * j + ((j >> 2) << 3) + 8 * h;
}

__global__ void attn_f16wmma_kernel(const float* __restrict__ qg,
                                    const float* __restrict__ kg,
                                    const float* __restrict__ vg,
                                    const uint8_t* __restrict__ maskg,
                                    float* __restrict__ outg,   // [B][L][64]
                                    float* __restrict__ attng)  // [B][L][L]
{
    extern __shared__ __align__(16) char smem[];
    _Float16* p_lds   = (_Float16*)smem;                            // [16][P_STRIDE]
    _Float16* q_lds   = (_Float16*)(smem + SMEM_P);                 // [16][Q_STRIDE]
    float*    o_lds   = (float*)(smem + SMEM_P + SMEM_Q);           // [2][16][64]
    float*    inv_lds = (float*)(smem + SMEM_P + SMEM_Q + SMEM_O);  // [16]

    const int b    = blockIdx.y;
    const int q0   = blockIdx.x * 16;
    const int tid  = threadIdx.x;
    const int wave = tid >> 5;
    const int lane = tid & 31;
    const int ln16 = lane & 15;
    const int h    = lane >> 4;

    // ---------- load Q tile (16x64) fp32 -> f16 LDS ----------
    for (int i = tid; i < 16 * 64; i += 256) {
        const int m = i >> 6, d = i & 63;
        q_lds[m * Q_STRIDE + d] = (_Float16)qg[((size_t)b * L_ + q0 + m) * D_ + d];
    }
    __syncthreads();

    // ---------- Q A-fragments, reused for all 128 key tiles ----------
    v16h a0, a1;
    {
        const _Float16* qr = &q_lds[ln16 * Q_STRIDE];
        #pragma unroll
        for (int j = 0; j < 8; ++j) {
            const int d = frag_d(j, h);
            const v2h q0p = *(const v2h*)(qr + d);
            const v2h q1p = *(const v2h*)(qr + 32 + d);
            a0[2 * j] = q0p[0]; a0[2 * j + 1] = q0p[1];
            a1[2 * j] = q1p[0]; a1[2 * j + 1] = q1p[1];
        }
    }

    // ---------- S = (Q K^T) * scale -> f16 LDS; wave w owns key tiles w, w+8, ... ----------
    for (int kt = wave; kt < L_ / 16; kt += 8) {
        const float* kr = &kg[((size_t)b * L_ + kt * 16 + ln16) * D_];
        __builtin_prefetch(kr + 8 * 16 * D_, 0, 0);   // next tile for this wave
        v16h b0, b1;
        #pragma unroll
        for (int j = 0; j < 8; ++j) {
            const int d = frag_d(j, h);
            const float2 p0 = *(const float2*)(kr + d);
            const float2 p1 = *(const float2*)(kr + 32 + d);
            b0[2 * j]     = (_Float16)p0.x;
            b0[2 * j + 1] = (_Float16)p0.y;
            b1[2 * j]     = (_Float16)p1.x;
            b1[2 * j + 1] = (_Float16)p1.y;
        }
        v8f c = {};
        c = __builtin_amdgcn_wmma_f32_16x16x32_f16(false, a0, false, b0, (short)0, c, false, false);
        c = __builtin_amdgcn_wmma_f32_16x16x32_f16(false, a1, false, b1, (short)0, c, false, false);

        const int col = kt * 16 + ln16;
        #pragma unroll
        for (int vv = 0; vv < 8; ++vv) {
            const int m = vv + 8 * h;   // C/D layout: M = vgpr + 8*(lane/16), N = lane%16
            p_lds[m * P_STRIDE + col] = (_Float16)(c[vv] * SCALE);
        }
    }
    __syncthreads();

    // ---------- masked softmax over keys: wave w handles rows 2w, 2w+1 ----------
    // 4 columns per lane: coalesced packed-u32 mask loads, v4h LDS, v4f attn stores
    const size_t attn_row_base = (size_t)b * L_ + q0;
    #pragma unroll
    for (int rr = 0; rr < 2; ++rr) {
        const int r = 2 * wave + rr;
        _Float16* prow = &p_lds[r * P_STRIDE];
        const uint32_t* mrow = (const uint32_t*)(maskg + (attn_row_base + r) * L_);

        // pass 1: apply mask fill, find row max
        float mx = -3.4e38f;
        for (int it = 0; it < 16; ++it) {
            const int c4 = it * 32 + lane;            // unit = 4 columns
            const uint32_t mw = __builtin_nontemporal_load(mrow + c4);
            v4h sv = *(v4h*)(prow + c4 * 4);
            #pragma unroll
            for (int i = 0; i < 4; ++i) {
                if ((mw >> (8 * i)) & 0xFFu) sv[i] = MASKH;
                mx = fmaxf(mx, (float)sv[i]);
            }
            *(v4h*)(prow + c4 * 4) = sv;
        }
        #pragma unroll
        for (int off = 16; off > 0; off >>= 1) mx = fmaxf(mx, __shfl_xor(mx, off, 32));

        // pass 2: exponentiate (store unnormalized e back), accumulate sum
        float sum = 0.f;
        for (int it = 0; it < 16; ++it) {
            const int c4 = it * 32 + lane;
            v4h sv = *(v4h*)(prow + c4 * 4);
            v4h ev;
            #pragma unroll
            for (int i = 0; i < 4; ++i) {
                const float e = __expf((float)sv[i] - mx);
                ev[i] = (_Float16)e;
                sum += e;
            }
            *(v4h*)(prow + c4 * 4) = ev;
        }
        #pragma unroll
        for (int off = 16; off > 0; off >>= 1) sum += __shfl_xor(sum, off, 32);
        const float inv = 1.f / sum;
        if (lane == 0) inv_lds[r] = inv;

        // pass 3: stream normalized fp32 probabilities to global (write-once -> NT)
        v4f* arow = (v4f*)(attng + (attn_row_base + r) * L_);
        for (int it = 0; it < 16; ++it) {
            const int c4 = it * 32 + lane;
            const v4h ev = *(v4h*)(prow + c4 * 4);
            v4f o;
            o[0] = (float)ev[0] * inv;
            o[1] = (float)ev[1] * inv;
            o[2] = (float)ev[2] * inv;
            o[3] = (float)ev[3] * inv;
            __builtin_nontemporal_store(o, arow + c4);
        }
    }
    __syncthreads();

    // ---------- O = E @ V (unnormalized), 4 N-tiles x split-K 2 across 8 waves ----------
    const int nt   = wave & 3;
    const int half = wave >> 2;
    v8f acc = {};
    for (int chunk = half; chunk < L_ / 32; chunk += 2) {
        const int kbase = chunk * 32;
        const _Float16* pr = &p_lds[ln16 * P_STRIDE + kbase];
        const float*    vb = &vg[((size_t)b * L_ + kbase) * D_ + nt * 16 + ln16];
        v16h aa, bb;
        #pragma unroll
        for (int j = 0; j < 8; ++j) {
            const int d = frag_d(j, h);
            const v2h pp = *(const v2h*)(pr + d);
            aa[2 * j]     = pp[0];
            aa[2 * j + 1] = pp[1];
            bb[2 * j]     = (_Float16)vb[(size_t)d * D_];
            bb[2 * j + 1] = (_Float16)vb[(size_t)(d + 1) * D_];
        }
        acc = __builtin_amdgcn_wmma_f32_16x16x32_f16(false, aa, false, bb, (short)0, acc, false, false);
    }
    #pragma unroll
    for (int vv = 0; vv < 8; ++vv) {
        const int m = vv + 8 * h;
        o_lds[(half * 16 + m) * 64 + nt * 16 + ln16] = acc[vv];
    }
    __syncthreads();

    // ---------- combine split-K partials, apply 1/rowsum, write O ----------
    for (int i = tid; i < 16 * 64; i += 256) {
        const int m = i >> 6, n = i & 63;
        outg[((size_t)b * L_ + q0 + m) * D_ + n] =
            (o_lds[m * 64 + n] + o_lds[(16 + m) * 64 + n]) * inv_lds[m];
    }
}

extern "C" void kernel_launch(void* const* d_in, const int* in_sizes, int n_in,
                              void* d_out, int out_size, void* d_ws, size_t ws_size,
                              hipStream_t stream) {
    (void)in_sizes; (void)n_in; (void)out_size; (void)d_ws; (void)ws_size;
    const float*   q    = (const float*)d_in[0];
    const float*   k    = (const float*)d_in[1];
    const float*   v    = (const float*)d_in[2];
    const uint8_t* mask = (const uint8_t*)d_in[3];   // jnp.bool_ -> 1 byte/elem

    float* out  = (float*)d_out;                      // tuple order: output first
    float* attn = out + (size_t)B_ * L_ * D_;         // then full attn matrix

    static_assert(SMEM_TOTAL <= 160 * 1024, "allow >=2 workgroups per WGP");
    (void)hipFuncSetAttribute((const void*)attn_f16wmma_kernel,
                              hipFuncAttributeMaxDynamicSharedMemorySize, SMEM_TOTAL);

    dim3 grid(L_ / 16, B_);   // 128 query tiles x 32 batches
    attn_f16wmma_kernel<<<grid, 256, SMEM_TOTAL, stream>>>(q, k, v, mask, out, attn);
}
